// MultiheadSelfAttention_14851996909848
// MI455X (gfx1250) — matrix operand order repaired and checked by reading, not compiled, not run on hardware
//
#include <hip/hip_runtime.h>
#include <hip/hip_bf16.h>
#include <math.h>

typedef _Float16 v8h  __attribute__((ext_vector_type(8)));
typedef _Float16 v16h __attribute__((ext_vector_type(16)));
typedef float    v8f  __attribute__((ext_vector_type(8)));

#define BB  4
#define SS  2048
#define DD  1024
#define HH  16
#define DKK 64

// ---- fragment loaders -------------------------------------------------------
// A fragment (16x32, f16): lane holds row m=lane&15, k in {kb..kb+7, kb+16..kb+23},
// kb = (lane<16)?0:8.  Two contiguous 16B loads.
static __device__ __forceinline__ v16h load_a_frag_f16(const _Float16* base, int ld, int lane) {
  const int m  = lane & 15;
  const int kb = (lane < 16) ? 0 : 8;
  const _Float16* p = base + (size_t)m * ld + kb;
  v8h lo = *(const v8h*)(p);
  v8h hi = *(const v8h*)(p + 16);
  v16h a;
#pragma unroll
  for (int i = 0; i < 8; ++i) { a[i] = lo[i]; a[8 + i] = hi[i]; }
  return a;
}

// B fragment (32x16, f16) from an n-major (k-contiguous) tile: lane holds
// column n=lane&15, k in {kb..kb+7, kb+16..kb+23}, kb = (lane<16)?0:8.  Two contiguous 16B loads.
static __device__ __forceinline__ v16h load_b_frag_f16(const _Float16* baseT, int ld, int lane) {
  const int n  = lane & 15;
  const int kb = (lane < 16) ? 0 : 8;
  const _Float16* p = baseT + (size_t)n * ld + kb;
  v8h lo = *(const v8h*)(p);
  v8h hi = *(const v8h*)(p + 16);
  v16h b;
#pragma unroll
  for (int i = 0; i < 8; ++i) { b[i] = lo[i]; b[8 + i] = hi[i]; }
  return b;
}

// ---- kernel 0: bulk f32 -> f16 conversion (x and weights) -------------------
__global__ __launch_bounds__(256)
void cvt_f16_kernel(const float* __restrict__ src, _Float16* __restrict__ dst, int n8) {
  const int i = blockIdx.x * 256 + threadIdx.x;
  if (i >= n8) return;
  const v8f v = *(const v8f*)(src + (size_t)i * 8);
  v8h h;
#pragma unroll
  for (int j = 0; j < 8; ++j) h[j] = (_Float16)v[j];
  *(v8h*)(dst + (size_t)i * 8) = h;
}

// ---- kernel 1: fused QKV projection + RoPE (+ V transpose), 32x64 tile ------
__global__ __launch_bounds__(32)
void qkv_rope_kernel(const _Float16* __restrict__ xh,
                     const _Float16* __restrict__ Whq,
                     const _Float16* __restrict__ Whk,
                     const _Float16* __restrict__ Whv,
                     const int*      __restrict__ pos,
                     _Float16* __restrict__ Qr,
                     _Float16* __restrict__ Kr,
                     _Float16* __restrict__ Vt) {
  __shared__ float tile[32][DKK];               // 32 rows x one full head (64), f32
  const int lane  = threadIdx.x;
  const int m0    = blockIdx.x * 32;            // row block in flattened (B*S)
  const int h     = blockIdx.y;                 // head index == 64-col block
  const int which = blockIdx.z;                 // 0=Q 1=K 2=V
  const _Float16* W = (which == 0) ? Whq : (which == 1) ? Whk : Whv;
  const int n0    = h * DKK;

  v8f acc[2][4] = {{{}, {}, {}, {}}, {{}, {}, {}, {}}};
  for (int kk = 0; kk < DD; kk += 32) {
    v16h a0 = load_a_frag_f16(xh + (size_t)m0 * DD + kk,        DD, lane);
    v16h a1 = load_a_frag_f16(xh + (size_t)(m0 + 16) * DD + kk, DD, lane);
#pragma unroll
    for (int t = 0; t < 4; ++t) {
      v16h b = load_b_frag_f16(W + (size_t)(n0 + t * 16) * DD + kk, DD, lane);
      acc[0][t] = __builtin_amdgcn_wmma_f32_16x16x32_f16(false, a0, false, b,
                                                         (short)0, acc[0][t], false, false);
      acc[1][t] = __builtin_amdgcn_wmma_f32_16x16x32_f16(false, a1, false, b,
                                                         (short)0, acc[1][t], false, false);
    }
  }

  // C fragments -> LDS (row m = sub*16 + r + 8*(lane>=16), col n = t*16 + (lane&15))
  {
    const int mb = (lane < 16) ? 0 : 8;
    const int n  = lane & 15;
#pragma unroll
    for (int sub = 0; sub < 2; ++sub)
#pragma unroll
      for (int t = 0; t < 4; ++t)
#pragma unroll
        for (int r = 0; r < 8; ++r)
          tile[sub * 16 + mb + r][t * 16 + n] = acc[sub][t][r];
  }
  __syncthreads();

  const int b  = m0 / SS;
  const int s0 = m0 % SS;

  if (which < 2) {
    // RoPE in f32: lane owns pair column j=lane (dims 2j, 2j+1) for all 32 rows
    const float inv = expf(-9.210340372f * ((float)lane / 32.0f));   // theta^(-2j/64)
#pragma unroll
    for (int i = 0; i < 32; ++i) {
      float p   = (float)pos[s0 + i];
      float ang = p * inv;
      float c = cosf(ang), sn = sinf(ang);
      float x1 = tile[i][2 * lane], x2 = tile[i][2 * lane + 1];
      tile[i][2 * lane]     = x1 * c - x2 * sn;
      tile[i][2 * lane + 1] = x1 * sn + x2 * c;
    }
    __syncthreads();
    _Float16* dst = (which == 0) ? Qr : Kr;     // (B,H,S,DK)
    size_t base = (((size_t)b * HH + h) * SS + (s0 + lane)) * DKK;   // one row per lane
#pragma unroll
    for (int i = 0; i < DKK; ++i) dst[base + i] = (_Float16)tile[lane][i];
  } else {
    // V stored transposed: Vt (B,H,DK,S) so P·V B-fragments are k-contiguous
#pragma unroll
    for (int half = 0; half < 2; ++half) {
      const int d = lane + half * 32;
      size_t base = (((size_t)b * HH + h) * DKK + d) * SS + s0;
#pragma unroll
      for (int i = 0; i < 32; ++i) Vt[base + i] = (_Float16)tile[i][d];
    }
  }
}

// ---- kernel 2: causal flash attention, one wave per 16-query tile -----------
__global__ __launch_bounds__(32)
void attn_kernel(const _Float16* __restrict__ Qr,
                 const _Float16* __restrict__ Kr,
                 const _Float16* __restrict__ Vt,
                 _Float16* __restrict__ AO) {
  __shared__ _Float16 Pt[16][32];               // probability tile (C->A re-layout)
  __shared__ _Float16 Ot[16][DKK];              // output staging
  const int lane = threadIdx.x;
  const int q0 = blockIdx.x * 16;
  const int h  = blockIdx.y;
  const int b  = blockIdx.z;
  const size_t bh = (size_t)b * HH + h;
  const _Float16* Qb = Qr + bh * SS * DKK;
  const _Float16* Kb = Kr + bh * SS * DKK;
  const _Float16* Vb = Vt + bh * DKK * SS;

  const v16h qa0 = load_a_frag_f16(Qb + (size_t)q0 * DKK + 0,  DKK, lane);
  const v16h qa1 = load_a_frag_f16(Qb + (size_t)q0 * DKK + 32, DKK, lane);

  v8f o[4] = {{}, {}, {}, {}};
  float rmax[8], rsum[8];
#pragma unroll
  for (int r = 0; r < 8; ++r) { rmax[r] = -INFINITY; rsum[r] = 0.0f; }

  const int hi  = (lane >= 16) ? 1 : 0;
  const int nlo = lane & 15;
  const float scale = 0.125f;                   // 1/sqrt(64)

  for (int j = 0; j < q0 + 16; j += 32) {       // causal loop bound
    v8f st[2];
#pragma unroll
    for (int t = 0; t < 2; ++t) {
      v16h kb0 = load_b_frag_f16(Kb + (size_t)(j + t * 16) * DKK + 0,  DKK, lane);
      v16h kb1 = load_b_frag_f16(Kb + (size_t)(j + t * 16) * DKK + 32, DKK, lane);
      v8f s = {};
      s = __builtin_amdgcn_wmma_f32_16x16x32_f16(false, qa0, false, kb0, (short)0, s, false, false);
      s = __builtin_amdgcn_wmma_f32_16x16x32_f16(false, qa1, false, kb1, (short)0, s, false, false);
      st[t] = s;
    }
    // online softmax, per row r (lanes 0-15 hold row r, lanes 16-31 row r+8)
#pragma unroll
    for (int r = 0; r < 8; ++r) {
      const int q = q0 + r + 8 * hi;
      const float s0 = (j + nlo      <= q) ? st[0][r] * scale : -INFINITY;
      const float s1 = (j + 16 + nlo <= q) ? st[1][r] * scale : -INFINITY;
      float v = fmaxf(s0, s1);
      v = fmaxf(v, __shfl_xor(v, 1, 32));
      v = fmaxf(v, __shfl_xor(v, 2, 32));
      v = fmaxf(v, __shfl_xor(v, 4, 32));
      v = fmaxf(v, __shfl_xor(v, 8, 32));
      const float nm   = fmaxf(rmax[r], v);
      const float corr = __expf(rmax[r] - nm);
      const float p0 = __expf(s0 - nm);
      const float p1 = __expf(s1 - nm);
      float ps = p0 + p1;
      ps += __shfl_xor(ps, 1, 32);
      ps += __shfl_xor(ps, 2, 32);
      ps += __shfl_xor(ps, 4, 32);
      ps += __shfl_xor(ps, 8, 32);
      rsum[r] = rsum[r] * corr + ps;
      rmax[r] = nm;
#pragma unroll
      for (int t2 = 0; t2 < 4; ++t2) o[t2][r] *= corr;
      Pt[r + 8 * hi][nlo]      = (_Float16)p0;
      Pt[r + 8 * hi][16 + nlo] = (_Float16)p1;
    }
    __syncthreads();
    const v16h pa = load_a_frag_f16(&Pt[0][0], 32, lane);
#pragma unroll
    for (int t2 = 0; t2 < 4; ++t2) {
      v16h vb = load_b_frag_f16(Vb + (size_t)(t2 * 16) * SS + j, SS, lane);
      o[t2] = __builtin_amdgcn_wmma_f32_16x16x32_f16(false, pa, false, vb,
                                                     (short)0, o[t2], false, false);
    }
    __syncthreads();                            // Pt reused next iteration
  }

#pragma unroll
  for (int t2 = 0; t2 < 4; ++t2)
#pragma unroll
    for (int r = 0; r < 8; ++r)
      Ot[r + 8 * hi][t2 * 16 + nlo] = (_Float16)(o[t2][r] / rsum[r]);
  __syncthreads();

  // AO layout (B,S,H*DK) == (B,S,D): ready for the output projection GEMM
  const int row = lane >> 1, cb = (lane & 1) * 32;
  size_t base = ((size_t)b * SS + q0 + row) * DD + (size_t)h * DKK + cb;
#pragma unroll
  for (int i = 0; i < 32; ++i) AO[base + i] = Ot[row][cb + i];
}

// ---- kernel 3: output projection (AO @ Wo^T), 32x64 tile, f32 result --------
__global__ __launch_bounds__(32)
void oproj_kernel(const _Float16* __restrict__ AO,
                  const _Float16* __restrict__ Who,
                  float* __restrict__ out) {
  __shared__ float tile[32][DKK];
  const int lane = threadIdx.x;
  const int m0 = blockIdx.x * 32;
  const int n0 = blockIdx.y * DKK;

  v8f acc[2][4] = {{{}, {}, {}, {}}, {{}, {}, {}, {}}};
  for (int kk = 0; kk < DD; kk += 32) {
    v16h a0 = load_a_frag_f16(AO + (size_t)m0 * DD + kk,        DD, lane);
    v16h a1 = load_a_frag_f16(AO + (size_t)(m0 + 16) * DD + kk, DD, lane);
#pragma unroll
    for (int t = 0; t < 4; ++t) {
      v16h b = load_b_frag_f16(Who + (size_t)(n0 + t * 16) * DD + kk, DD, lane);
      acc[0][t] = __builtin_amdgcn_wmma_f32_16x16x32_f16(false, a0, false, b,
                                                         (short)0, acc[0][t], false, false);
      acc[1][t] = __builtin_amdgcn_wmma_f32_16x16x32_f16(false, a1, false, b,
                                                         (short)0, acc[1][t], false, false);
    }
  }
  const int mb = (lane < 16) ? 0 : 8;
  const int n  = lane & 15;
#pragma unroll
  for (int sub = 0; sub < 2; ++sub)
#pragma unroll
    for (int t = 0; t < 4; ++t)
#pragma unroll
      for (int r = 0; r < 8; ++r)
        tile[sub * 16 + mb + r][t * 16 + n] = acc[sub][t][r];
  __syncthreads();

  size_t base = ((size_t)m0 + lane) * DD + n0;  // one row per lane
#pragma unroll
  for (int i = 0; i < DKK; ++i) out[base + i] = tile[lane][i];
}

// ---- launcher ---------------------------------------------------------------
extern "C" void kernel_launch(void* const* d_in, const int* in_sizes, int n_in,
                              void* d_out, int out_size, void* d_ws, size_t ws_size,
                              hipStream_t stream) {
  const float* x  = (const float*)d_in[0];
  const float* Wq = (const float*)d_in[1];
  const float* Wk = (const float*)d_in[2];
  const float* Wv = (const float*)d_in[3];
  const float* Wo = (const float*)d_in[4];
  const int*  pos = (const int*)d_in[5];
  float* out = (float*)d_out;

  // workspace layout (f16): Qr,Kr (B,H,S,DK); Vt (B,H,DK,S); AO (B,S,D);
  // xh (B,S,D); Whq/Whk/Whv/Who (D,D)  -> ~92 MB total
  const size_t qkv_elems = (size_t)BB * HH * SS * DKK;   // 8388608
  const size_t x_elems   = (size_t)BB * SS * DD;         // 8388608
  const size_t w_elems   = (size_t)DD * DD;              // 1048576
  _Float16* Qr  = (_Float16*)d_ws;
  _Float16* Kr  = Qr  + qkv_elems;
  _Float16* Vt  = Kr  + qkv_elems;
  _Float16* AO  = Vt  + qkv_elems;
  _Float16* xh  = AO  + qkv_elems;
  _Float16* Whq = xh  + x_elems;
  _Float16* Whk = Whq + w_elems;
  _Float16* Whv = Whk + w_elems;
  _Float16* Who = Whv + w_elems;

  // 0) convert activations + weights to f16 once
  cvt_f16_kernel<<<(int)(x_elems / 8 / 256), 256, 0, stream>>>(x,  xh,  (int)(x_elems / 8));
  cvt_f16_kernel<<<(int)(w_elems / 8 / 256), 256, 0, stream>>>(Wq, Whq, (int)(w_elems / 8));
  cvt_f16_kernel<<<(int)(w_elems / 8 / 256), 256, 0, stream>>>(Wk, Whk, (int)(w_elems / 8));
  cvt_f16_kernel<<<(int)(w_elems / 8 / 256), 256, 0, stream>>>(Wv, Whv, (int)(w_elems / 8));
  cvt_f16_kernel<<<(int)(w_elems / 8 / 256), 256, 0, stream>>>(Wo, Who, (int)(w_elems / 8));

  // 1) QKV projection + RoPE (+ V transpose)
  dim3 g1((BB * SS) / 32, DD / DKK, 3);
  qkv_rope_kernel<<<g1, 32, 0, stream>>>(xh, Whq, Whk, Whv, pos, Qr, Kr, Vt);

  // 2) causal flash attention
  dim3 g2(SS / 16, HH, BB);
  attn_kernel<<<g2, 32, 0, stream>>>(Qr, Kr, Vt, AO);

  // 3) output projection
  dim3 g3((BB * SS) / 32, DD / DKK);
  oproj_kernel<<<g3, 32, 0, stream>>>(AO, Who, out);
}
